// SelfAttention_43482248905221
// MI455X (gfx1250) — compile-verified, matrix operand order checked by reading, not run on hardware
//
#include <hip/hip_runtime.h>
#include <hip/hip_bf16.h>

// ---------------------------------------------------------------------------
// CDNA5 (gfx1250) self-attention:
//   proj (f32->bf16 WMMA GEMM) -> scores (bf16 WMMA) -> softmax -> P@V (WMMA)
// All GEMMs compute from double-buffered bf16 LDS tiles.  bf16 inputs are
// staged with global_load_async_to_lds_b128 (ASYNCcnt); f32 inputs are
// converted to bf16 once while staging (load -> v_cvt_pk_bf16_f32 -> ds_store),
// so the inner loop is always ds_load_b128 -> v_wmma_f32_16x16x32_bf16.
// ---------------------------------------------------------------------------

typedef __attribute__((ext_vector_type(16))) __bf16 v16bf;
typedef __attribute__((ext_vector_type(8)))  float  v8f;

__device__ __forceinline__ void wait_asynccnt0() {
  asm volatile("s_wait_asynccnt 0x0" ::: "memory");
}

__device__ __forceinline__ void async_ld_b128(unsigned lds_addr, const void* gaddr) {
  // GLOBAL_LOAD_ASYNC_TO_LDS_B128: per-lane 16B memory -> LDS, tracked by ASYNCcnt.
  asm volatile("global_load_async_to_lds_b128 %0, %1, off"
               :: "v"(lds_addr), "v"(gaddr) : "memory");
}

// Build a 16x32 bf16 WMMA fragment from an LDS tile row.
// p points at (row, k = g*8); per-lane chunks are [k..k+8) and [k+16..k+24).
__device__ __forceinline__ v16bf frag_bf16_lds(const unsigned char* p) {
  union { uint4 u[2]; v16bf v; } u;
  u.u[0] = *(const uint4*)p;           // 8 bf16
  u.u[1] = *(const uint4*)(p + 32);    // 8 bf16, 16 elements later
  return u.v;
}

// Batched NT GEMM: C[z][m][n] = alpha * sum_k A[z][m][k]*B[z][n][k] (+bias[n]).
// 256 threads = 8 waves.  Block tile 128(M) x 128(N); wave = 16(M) x 128(N)
// (8 accumulators, A-fragment reused 8x).  K stepped by 32.
// LDS: 2 buffers x (128 A rows + 128 B rows) x padded 32-bf16 row (80B).
// Strides in elements of the input type (f32 if IN_F32 else bf16).
template<bool IN_F32, bool OUT_BF16, bool OUT_TRANS, bool BIAS>
__global__ __launch_bounds__(256)
void gemm_nt_wmma(const void* __restrict__ Ap, const void* __restrict__ Bp,
                  const float* __restrict__ bias, void* __restrict__ Cp,
                  int K, int lda, int ldb, int ldc,
                  long long sA, long long sB, long long sC, float alpha)
{
  constexpr unsigned RS    = 32 * 2 + 16;   // padded bf16 LDS row stride (80B)
  constexpr unsigned BUFSZ = 256u * RS;     // one buffer: 128 A + 128 B rows

  __shared__ __align__(16) unsigned char smem[2 * BUFSZ];

  const int tid  = threadIdx.x;
  const int lane = tid & 31;
  const int wave = tid >> 5;
  const int r    = lane & 15;
  const int g    = lane >> 4;
  const int mblk = blockIdx.x * 128;
  const int nblk = blockIdx.y * 128;

  const long long abase = (long long)blockIdx.z * sA;
  const long long bbase = (long long)blockIdx.z * sB;
  const long long cbase = (long long)blockIdx.z * sC;

  const unsigned ldsbase = (unsigned)(uintptr_t)&smem[0];

  // Cooperative fill of one double-buffer slot for K slice [kk, kk+32).
  auto issue_tiles = [&](int buf, int kk) {
    if constexpr (!IN_F32) {
      // bf16 source: raw async copy, 4 x 16B chunks per thread.
      const unsigned bb = ldsbase + (unsigned)buf * BUFSZ;
      #pragma unroll
      for (int c = 0; c < 4; ++c) {
        const int idx = tid + c * 256;       // 0..1023
        const int row = idx >> 2;            // 0..127 = A rows, 128..255 = B rows
        const int kc  = (idx & 3) * 16;      // byte offset inside 64B row span
        const char* gsrc;
        if (row < 128)
          gsrc = (const char*)Ap +
                 (abase + (long long)(mblk + row) * lda + kk) * 2 + kc;
        else
          gsrc = (const char*)Bp +
                 (bbase + (long long)(nblk + (row - 128)) * ldb + kk) * 2 + kc;
        async_ld_b128(bb + (unsigned)row * RS + (unsigned)kc, gsrc);
      }
    } else {
      // f32 source: load 4 floats, convert to bf16 once, ds_store_b64.
      unsigned char* bb = &smem[(unsigned)buf * BUFSZ];
      #pragma unroll
      for (int c = 0; c < 8; ++c) {
        const int idx = tid + c * 256;       // 0..2047
        const int row = idx >> 3;            // 0..127 = A rows, 128..255 = B rows
        const int kc  = (idx & 7) * 4;       // f32 element offset inside 32
        const float* gsrc;
        if (row < 128)
          gsrc = (const float*)Ap + abase + (long long)(mblk + row) * lda + kk + kc;
        else
          gsrc = (const float*)Bp + bbase + (long long)(nblk + (row - 128)) * ldb + kk + kc;
        float4 f = *(const float4*)gsrc;
        union { __bf16 h[4]; uint2 u; } cv;
        cv.h[0] = (__bf16)f.x; cv.h[1] = (__bf16)f.y;
        cv.h[2] = (__bf16)f.z; cv.h[3] = (__bf16)f.w;
        *(uint2*)(bb + (unsigned)row * RS + (unsigned)(kc * 2)) = cv.u;
      }
    }
  };

  v8f acc[8];
  #pragma unroll
  for (int t = 0; t < 8; ++t)
    #pragma unroll
    for (int j = 0; j < 8; ++j) acc[t][j] = 0.0f;

  issue_tiles(0, 0);
  int cur = 0;

  for (int k = 0; k < K; k += 32) {
    if constexpr (!IN_F32) wait_asynccnt0();
    __syncthreads();                       // tiles for `cur` visible to all waves
    if (k + 32 < K) issue_tiles(cur ^ 1, k + 32);   // prefetch next slice

    const unsigned char* bufp = &smem[(unsigned)cur * BUFSZ];
    const unsigned aoff = (unsigned)(wave * 16 + r) * RS + (unsigned)(g * 16);

    v16bf a = frag_bf16_lds(bufp + aoff);

    // Software pipeline: fetch B(t+1) from LDS while WMMA(t) issues.
    v16bf bcur = frag_bf16_lds(bufp + (unsigned)(128 + r) * RS + (unsigned)(g * 16));
    #pragma unroll
    for (int t = 0; t < 8; ++t) {
      v16bf bnxt = bcur;
      if (t < 7)
        bnxt = frag_bf16_lds(bufp + (unsigned)(128 + (t + 1) * 16 + r) * RS +
                             (unsigned)(g * 16));
      acc[t] = __builtin_amdgcn_wmma_f32_16x16x32_bf16(
          false, a, false, bcur, (short)0, acc[t], false, false);
      bcur = bnxt;
    }
    cur ^= 1;
  }

  // C/D layout: element j of v8f -> row mblk + wave*16 + 8*g + j,
  //             col nblk + t*16 + (lane&15).
  #pragma unroll
  for (int t = 0; t < 8; ++t) {
    const int n = nblk + t * 16 + r;
    float badd = 0.0f;
    if constexpr (BIAS) badd = bias[n];
    if constexpr (OUT_TRANS) {
      // C^T[n][m]: 8 consecutive m -> one packed 16B store.
      union { __bf16 h[8]; uint4 u; } pk;
      #pragma unroll
      for (int j = 0; j < 8; ++j) pk.h[j] = (__bf16)(acc[t][j] * alpha + badd);
      const long long m0 = mblk + wave * 16 + 8 * g;
      *(uint4*)&((__bf16*)Cp)[cbase + (long long)n * ldc + m0] = pk.u;
    } else {
      #pragma unroll
      for (int j = 0; j < 8; ++j) {
        const int m = mblk + wave * 16 + 8 * g + j;
        const float v = acc[t][j] * alpha + badd;
        if constexpr (OUT_BF16)
          ((__bf16*)Cp)[cbase + (long long)m * ldc + n] = (__bf16)v;
        else
          ((float*)Cp)[cbase + (long long)m * ldc + n] = v;
      }
    }
  }
}

// Row softmax over f32 energies, writing bf16 probabilities IN PLACE at the
// start of each (byte-)row.  One 256-thread block per row (rowlen = 2048).
__global__ __launch_bounds__(256)
void softmax_rows_kernel(float* __restrict__ energies, int rowlen)
{
  __shared__ float red[8];
  float* row = energies + (long long)blockIdx.x * rowlen;
  const int tid = threadIdx.x;

  float vals[8];
  float m = -3.4e38f;
  #pragma unroll
  for (int i = 0; i < 8; ++i) {
    vals[i] = row[tid + (i << 8)];
    m = fmaxf(m, vals[i]);
  }
  #pragma unroll
  for (int off = 16; off >= 1; off >>= 1)
    m = fmaxf(m, __shfl_xor(m, off, 32));
  if ((tid & 31) == 0) red[tid >> 5] = m;
  __syncthreads();
  m = red[0];
  #pragma unroll
  for (int w = 1; w < 8; ++w) m = fmaxf(m, red[w]);
  __syncthreads();

  float s = 0.0f;
  #pragma unroll
  for (int i = 0; i < 8; ++i) {
    vals[i] = __expf(vals[i] - m);
    s += vals[i];
  }
  #pragma unroll
  for (int off = 16; off >= 1; off >>= 1)
    s += __shfl_xor(s, off, 32);
  if ((tid & 31) == 0) red[tid >> 5] = s;
  __syncthreads();
  s = 0.0f;
  #pragma unroll
  for (int w = 0; w < 8; ++w) s += red[w];
  const float inv = 1.0f / s;
  __syncthreads();   // every thread finished reading this row's f32 data

  __bf16* p = (__bf16*)row;   // in-place bf16 row (stride = 2*rowlen bf16 elems)
  #pragma unroll
  for (int i = 0; i < 8; ++i)
    p[tid + (i << 8)] = (__bf16)(vals[i] * inv);
}

// ---------------------------------------------------------------------------

extern "C" void kernel_launch(void* const* d_in, const int* in_sizes, int n_in,
                              void* d_out, int out_size, void* d_ws, size_t ws_size,
                              hipStream_t stream)
{
  (void)in_sizes; (void)n_in; (void)out_size; (void)ws_size;

  constexpr int B = 4, S = 2048, D = 1024;

  const float* q  = (const float*)d_in[0];
  const float* k  = (const float*)d_in[1];
  const float* v  = (const float*)d_in[2];
  const float* Wq = (const float*)d_in[3];
  const float* bq = (const float*)d_in[4];
  const float* Wk = (const float*)d_in[5];
  const float* bk = (const float*)d_in[6];
  const float* Wv = (const float*)d_in[7];
  const float* bv = (const float*)d_in[8];
  float* out = (float*)d_out;

  // Workspace layout:
  //   qp  bf16 [B][S][D], kp bf16 [B][S][D], vpT bf16 [B][D][S],
  //   E   f32  [B][S][S]  (softmax rewrites each row in place as bf16 P).
  __bf16* qp  = (__bf16*)d_ws;
  __bf16* kp  = qp  + (long long)B * S * D;
  __bf16* vpT = kp  + (long long)B * S * D;
  float*  E   = (float*)(vpT + (long long)B * S * D);

  dim3 blk(256);

  // --- Phase 1: projections (f32 inputs -> bf16 outputs), batched over z=B.
  {
    dim3 grid(S / 128, D / 128, B);
    gemm_nt_wmma<true, true, false, true><<<grid, blk, 0, stream>>>(
        q, Wq, bq, qp, /*K=*/D, /*lda=*/D, /*ldb=*/D, /*ldc=*/D,
        (long long)S * D, 0LL, (long long)S * D, 1.0f);
    gemm_nt_wmma<true, true, false, true><<<grid, blk, 0, stream>>>(
        k, Wk, bk, kp, D, D, D, D,
        (long long)S * D, 0LL, (long long)S * D, 1.0f);
    // vpT[n][m] = (v @ Wv^T + bv)[m][n]   (transposed store, ldc = S)
    gemm_nt_wmma<true, true, true, true><<<grid, blk, 0, stream>>>(
        v, Wv, bv, vpT, D, D, D, /*ldc=*/S,
        (long long)S * D, 0LL, (long long)D * S, 1.0f);
  }

  // --- Phase 2: energies E = qp @ kp^T / sqrt(D), f32 out.
  {
    dim3 grid(S / 128, S / 128, B);
    gemm_nt_wmma<false, false, false, false><<<grid, blk, 0, stream>>>(
        qp, kp, nullptr, E, /*K=*/D, D, D, /*ldc=*/S,
        (long long)S * D, (long long)S * D, (long long)S * S, 0.03125f);
  }

  // --- Phase 3: row softmax, P written bf16 in place (row stride 2*S bf16).
  softmax_rows_kernel<<<B * S, blk, 0, stream>>>(E, S);

  // --- Phase 4: out = P @ vpT^T  (A = bf16 P with lda = 2*S elements).
  {
    dim3 grid(S / 128, D / 128, B);
    gemm_nt_wmma<false, false, false, false><<<grid, blk, 0, stream>>>(
        (const __bf16*)E, vpT, nullptr, out, /*K=*/S,
        /*lda=*/2 * S, /*ldb=*/S, /*ldc=*/D,
        (long long)S * (2 * S), (long long)D * S, (long long)S * D, 1.0f);
  }
}
